// ReservoirComputingLayer_83537113907278
// MI455X (gfx1250) — matrix operand order repaired and checked by reading, not compiled, not run on hardware
//
#include <hip/hip_runtime.h>

// ---------------------------------------------------------------------------
// ReservoirComputingLayer for MI455X (gfx1250, wave32, WMMA)
//   out0: padded_state      (1024 x 4096) f32
//   out1: new_reservoir_wts (4096 x 4096) f32
// GEMM: state_pre = prev[:, :c] @ W[:c,:c] + inputs @ IW[:c,:]^T
// Matrix path: bf16x3 split-precision on V_WMMA_F32_16X16X32_BF16
//   x = hi + lo (bf16),  x*y ~= hi*hi + hi*lo + lo*hi  (~fp32 accuracy)
// ---------------------------------------------------------------------------

typedef __attribute__((ext_vector_type(16))) __bf16 v16bf;
typedef __attribute__((ext_vector_type(8)))  __bf16 v8bf;
typedef __attribute__((ext_vector_type(8)))  float  v8f;

#define B_DIM   1024
#define MAX_DIM 4096
#define DIN_DIM 512
#define LEAK    0.1f
#define THR     0.5f
#define RATE    0.01f
#define PRUNE   0.05f

#define M_TILE 64
#define N_TILE 64
#define K_TILE 32
#define LDK    40     // K_TILE + 8 pad: rows stay 16B-aligned, banks spread

// Split x into bf16 hi/lo and store 8 contiguous elements as one b128 each.
__device__ inline void split_store(__bf16* __restrict__ hp,
                                   __bf16* __restrict__ lp,
                                   const float* __restrict__ x) {
    v8bf h, l;
#pragma unroll
    for (int i = 0; i < 8; ++i) {
        __bf16 hb = (__bf16)x[i];
        h[i] = hb;
        l[i] = (__bf16)(x[i] - (float)hb);
    }
    *(v8bf*)hp = h;
    *(v8bf*)lp = l;
}

// Assemble a v16bf fragment from two contiguous 16-byte LDS chunks.
__device__ inline v16bf frag16(const __bf16* p0, const __bf16* p1) {
    v8bf a = *(const v8bf*)p0;
    v8bf b = *(const v8bf*)p1;
    return __builtin_shufflevector(a, b, 0, 1, 2, 3, 4, 5, 6, 7,
                                         8, 9, 10, 11, 12, 13, 14, 15);
}

__global__ __launch_bounds__(256)
void reservoir_state_kernel(const float* __restrict__ inputs,
                            const float* __restrict__ prev_state,
                            const float* __restrict__ W,
                            const float* __restrict__ IW,
                            const int*   __restrict__ cptr,
                            float*       __restrict__ out_state) {
    const int c = cptr[0];

    __shared__ __bf16 Ah[M_TILE][LDK];   // A tile hi: 64 rows x 32 k
    __shared__ __bf16 Al[M_TILE][LDK];   // A tile lo
    __shared__ __bf16 Bth[N_TILE][LDK];  // B tile hi, TRANSPOSED: [col][k]
    __shared__ __bf16 Btl[N_TILE][LDK];  // B tile lo, TRANSPOSED

    const int tid  = threadIdx.x;
    const int lane = tid & 31;
    const int wave = tid >> 5;
    const int wm   = wave & 3;    // M sub-tile 0..3
    const int wn   = wave >> 2;   // N sub-tile 0..1

    const int b0 = blockIdx.y * M_TILE;
    const int j0 = blockIdx.x * N_TILE;

    v8f acc0 = {};   // cols [wn*32 +  0 .. +15]
    v8f acc1 = {};   // cols [wn*32 + 16 .. +31]

    // 16-bit WMMA fragment lane mapping (ISA 7.12.2):
    //  A 16x32: lane L -> row L%16; K chunks [(L/16)*8 .. +7] and [16+(L/16)*8 .. +7]
    //  B 32x16: lane L -> col L%16; K chunks [(L/16)*8 .. +7] and [16+(L/16)*8 .. +7]
    const int arow = wm * 16 + (lane & 15);
    const int ac0  = (lane >> 4) * 8;
    const int kb   = (lane >> 4) * 16;
    const int colA = wn * 32 + (lane & 15);

    auto mma_tile = [&]() {
        v16bf ah  = frag16(&Ah[arow][ac0], &Ah[arow][16 + ac0]);
        v16bf al_ = frag16(&Al[arow][ac0], &Al[arow][16 + ac0]);

        v16bf b0h = frag16(&Bth[colA][ac0],      &Bth[colA][16 + ac0]);
        v16bf b0l = frag16(&Btl[colA][ac0],      &Btl[colA][16 + ac0]);
        v16bf b1h = frag16(&Bth[colA + 16][ac0], &Bth[colA + 16][16 + ac0]);
        v16bf b1l = frag16(&Btl[colA + 16][ac0], &Btl[colA + 16][16 + ac0]);

        acc0 = __builtin_amdgcn_wmma_f32_16x16x32_bf16(false, ah,  false, b0h,
                                                       (short)0, acc0, false, false);
        acc0 = __builtin_amdgcn_wmma_f32_16x16x32_bf16(false, ah,  false, b0l,
                                                       (short)0, acc0, false, false);
        acc0 = __builtin_amdgcn_wmma_f32_16x16x32_bf16(false, al_, false, b0h,
                                                       (short)0, acc0, false, false);

        acc1 = __builtin_amdgcn_wmma_f32_16x16x32_bf16(false, ah,  false, b1h,
                                                       (short)0, acc1, false, false);
        acc1 = __builtin_amdgcn_wmma_f32_16x16x32_bf16(false, ah,  false, b1l,
                                                       (short)0, acc1, false, false);
        acc1 = __builtin_amdgcn_wmma_f32_16x16x32_bf16(false, al_, false, b1h,
                                                       (short)0, acc1, false, false);
    };

    // ------------------ Phase 1: prev[:, :c] @ W[:c, :c] -------------------
    for (int k0 = 0; k0 < c; k0 += K_TILE) {
        // Stage A: prev 64x32 (thread t -> row t/4, k-octet (t%4)*8, contiguous)
        {
            const int r  = tid >> 2;
            const int kc = (tid & 3) * 8;
            const float* src = prev_state + (size_t)(b0 + r) * MAX_DIM + k0 + kc;
            float x[8];
            if (k0 + K_TILE <= c) {
                float4 u0 = *(const float4*)src;
                float4 u1 = *(const float4*)(src + 4);
                x[0] = u0.x; x[1] = u0.y; x[2] = u0.z; x[3] = u0.w;
                x[4] = u1.x; x[5] = u1.y; x[6] = u1.z; x[7] = u1.w;
            } else {
#pragma unroll
                for (int i = 0; i < 8; ++i) x[i] = (k0 + kc + i < c) ? src[i] : 0.0f;
            }
            split_store(&Ah[r][kc], &Al[r][kc], x);
        }
        // Stage B transposed: thread t -> col t%64, k-octet (t/64)*8.
        // Global reads: 64 consecutive columns per k-row (fully coalesced, L2-resident).
        // LDS writes: contiguous 16B per hi/lo array.
        {
            const int col = tid & 63;
            const int kg  = (tid >> 6) * 8;
            const float* src = W + (size_t)(k0 + kg) * MAX_DIM + (j0 + col);
            float x[8];
            if (k0 + K_TILE <= c && j0 + N_TILE <= c) {
#pragma unroll
                for (int i = 0; i < 8; ++i) x[i] = src[(size_t)i * MAX_DIM];
                if (k0 + K_TILE < c)  // prefetch next W tile -> global_prefetch_b8
                    __builtin_prefetch(W + (size_t)(k0 + K_TILE + kg) * MAX_DIM + (j0 + col), 0, 1);
            } else {
#pragma unroll
                for (int i = 0; i < 8; ++i)
                    x[i] = ((k0 + kg + i) < c && (j0 + col) < c) ? src[(size_t)i * MAX_DIM] : 0.0f;
            }
            split_store(&Bth[col][kg], &Btl[col][kg], x);
        }
        __syncthreads();
        mma_tile();
        __syncthreads();
    }

    // ------------------ Phase 2: inputs @ IW[:c, :]^T -----------------------
    for (int k0 = 0; k0 < DIN_DIM; k0 += K_TILE) {
        // Stage A: inputs 64x32 (always in range: DIN % 32 == 0)
        {
            const int r  = tid >> 2;
            const int kc = (tid & 3) * 8;
            const float* src = inputs + (size_t)(b0 + r) * DIN_DIM + k0 + kc;
            float4 u0 = *(const float4*)src;
            float4 u1 = *(const float4*)(src + 4);
            float x[8] = {u0.x, u0.y, u0.z, u0.w, u1.x, u1.y, u1.z, u1.w};
            split_store(&Ah[r][kc], &Al[r][kc], x);
        }
        // Stage B transposed: Bt[col][k] = IW[j0+col, k0+k] -- IW rows are
        // k-contiguous, so reads AND LDS writes are both contiguous.
        {
            const int col = tid >> 2;          // 0..63
            const int kc  = (tid & 3) * 8;
            const int gj  = j0 + col;
            float x[8];
            if (gj < c) {
                const float* src = IW + (size_t)gj * DIN_DIM + k0 + kc;
                float4 u0 = *(const float4*)src;
                float4 u1 = *(const float4*)(src + 4);
                x[0] = u0.x; x[1] = u0.y; x[2] = u0.z; x[3] = u0.w;
                x[4] = u1.x; x[5] = u1.y; x[6] = u1.z; x[7] = u1.w;
            } else {
#pragma unroll
                for (int i = 0; i < 8; ++i) x[i] = 0.0f;
            }
            split_store(&Bth[col][kc], &Btl[col][kc], x);
        }
        __syncthreads();
        mma_tile();
        __syncthreads();
    }

    // ------------------ Epilogue: leak + tanh + threshold + pad -------------
    // D layout: VGPR v -> M = v + (lane/16)*8 ; N = lane%16 (per sub-tile)
    const int m_base = wm * 16 + (lane >> 4) * 8;
    const int n_base = j0 + wn * 32 + (lane & 15);
#pragma unroll
    for (int v = 0; v < 8; ++v) {
        const int b = b0 + m_base + v;
        {
            const int j = n_base;
            const float pv = (j < c) ? prev_state[(size_t)b * MAX_DIM + j] : 0.0f;
            float s = (1.0f - LEAK) * pv + LEAK * tanhf(acc0[v]);
            s = (s > THR) ? (s - THR) : s;
            out_state[(size_t)b * MAX_DIM + j] = (j < c) ? s : 0.0f;
        }
        {
            const int j = n_base + 16;
            const float pv = (j < c) ? prev_state[(size_t)b * MAX_DIM + j] : 0.0f;
            float s = (1.0f - LEAK) * pv + LEAK * tanhf(acc1[v]);
            s = (s > THR) ? (s - THR) : s;
            out_state[(size_t)b * MAX_DIM + j] = (j < c) ? s : 0.0f;
        }
    }
}

// ---------------------------------------------------------------------------
// Weight grow/prune update (bandwidth-bound elementwise):
//   out[i,j] = (i<c && j<c) ? (|W[i,j]|<PRUNE ? 0 : W[i,j] + (grow_u[j]<RATE)*noise[i,j]) : 0
// grow broadcasts per COLUMN j.
// ---------------------------------------------------------------------------
__global__ __launch_bounds__(256)
void reservoir_weights_kernel(const float* __restrict__ W,
                              const float* __restrict__ grow_u,
                              const float* __restrict__ noise,
                              const int*   __restrict__ cptr,
                              float*       __restrict__ out_W) {
    const int c  = cptr[0];
    const int i  = blockIdx.y;
    const int j4 = (blockIdx.x * 256 + threadIdx.x) * 4;
    const size_t obase = (size_t)i * MAX_DIM + j4;

    float4 r;
    if (i < c && j4 + 3 < c) {
        float4 w = *(const float4*)(W + obase);  // W row stride 4096 -> 16B aligned
        const size_t nbase = (size_t)i * c + j4; // noise row stride c (runtime)
        const float n0 = noise[nbase + 0], n1 = noise[nbase + 1];
        const float n2 = noise[nbase + 2], n3 = noise[nbase + 3];
        const float g0 = (grow_u[j4 + 0] < RATE) ? 1.0f : 0.0f;
        const float g1 = (grow_u[j4 + 1] < RATE) ? 1.0f : 0.0f;
        const float g2 = (grow_u[j4 + 2] < RATE) ? 1.0f : 0.0f;
        const float g3 = (grow_u[j4 + 3] < RATE) ? 1.0f : 0.0f;
        r.x = (fabsf(w.x) < PRUNE) ? 0.0f : (w.x + g0 * n0);
        r.y = (fabsf(w.y) < PRUNE) ? 0.0f : (w.y + g1 * n1);
        r.z = (fabsf(w.z) < PRUNE) ? 0.0f : (w.z + g2 * n2);
        r.w = (fabsf(w.w) < PRUNE) ? 0.0f : (w.w + g3 * n3);
    } else {
        float t[4];
#pragma unroll
        for (int e = 0; e < 4; ++e) {
            const int j = j4 + e;
            float o = 0.0f;
            if (i < c && j < c) {
                const float w = W[(size_t)i * MAX_DIM + j];
                const float g = (grow_u[j] < RATE) ? 1.0f : 0.0f;
                const float n = noise[(size_t)i * c + j];
                o = (fabsf(w) < PRUNE) ? 0.0f : (w + g * n);
            }
            t[e] = o;
        }
        r.x = t[0]; r.y = t[1]; r.z = t[2]; r.w = t[3];
    }
    *(float4*)(out_W + obase) = r;
}

// ---------------------------------------------------------------------------
extern "C" void kernel_launch(void* const* d_in, const int* in_sizes, int n_in,
                              void* d_out, int out_size, void* d_ws, size_t ws_size,
                              hipStream_t stream) {
    const float* inputs = (const float*)d_in[0];   // (1024, 512)
    const float* prev   = (const float*)d_in[1];   // (1024, 4096)
    const float* W      = (const float*)d_in[2];   // (4096, 4096)
    const float* IW     = (const float*)d_in[3];   // (4096, 512)
    const float* grow_u = (const float*)d_in[4];   // (c,)
    const float* noise  = (const float*)d_in[5];   // (c, c)
    const int*   csz    = (const int*)d_in[6];     // scalar

    float* out_state = (float*)d_out;                        // (1024, 4096)
    float* out_W     = out_state + (size_t)B_DIM * MAX_DIM;  // (4096, 4096)

    dim3 gs(MAX_DIM / N_TILE, B_DIM / M_TILE);   // (64, 16)
    reservoir_state_kernel<<<gs, 256, 0, stream>>>(inputs, prev, W, IW, csz, out_state);

    dim3 gw(MAX_DIM / (256 * 4), MAX_DIM);       // (4, 4096)
    reservoir_weights_kernel<<<gw, 256, 0, stream>>>(W, grow_u, noise, csz, out_W);
}
